// GloAttnConv_34746285425029
// MI455X (gfx1250) — matrix operand order repaired and checked by reading, not compiled, not run on hardware
//
#include <hip/hip_runtime.h>

#define DI __device__ __forceinline__

typedef __attribute__((ext_vector_type(16))) __bf16 v16bf;
typedef __attribute__((ext_vector_type(8)))  float  v8f;

namespace {
constexpr int NN      = 100000;   // nodes
constexpr int CIN     = 256;      // in channels
constexpr int HD      = 256;      // heads * dhead (4*64)
constexpr int NTILES  = NN / 16;  // 6250  (NN % 16 == 0)
constexpr int NCHUNKS = NN / 32;  // 3125  (NN % 32 == 0)
constexpr int KVS_SPLIT = 50;     // split-K factor for kvs reduction

// ---- workspace byte offsets (total ~98.2 MB) ----
constexpr size_t KVS_OFF   = 0;                          // 65536 f32  kvs[4*64][256]
constexpr size_t KSSUM_OFF = 262144;                     // 256 f32
constexpr size_t VSSUM_OFF = 263168;                     // 256 f32
constexpr size_t WQ_OFF    = 264192;                     // 65536 bf16
constexpr size_t WK_OFF    = WQ_OFF + 131072;            // 65536 bf16 (contiguous after WQ)
constexpr size_t QS_OFF    = WK_OFF + 131072;            // NN*256 bf16 (51.2 MB)
constexpr size_t KS_OFF    = QS_OFF + (size_t)NN * HD * 2;
} // namespace

union BFr { v16bf v; uint4 q[2]; unsigned short s[16]; __bf16 b[16]; };
union V8  { v8f   v; float f[8]; };
union F4  { float4 v; float f[4]; };

// f32 -> bf16 via hardware convert (RNE)
DI unsigned short f2bf(float f) {
    union { __bf16 h; unsigned short s; } c;
    c.h = (__bf16)f;
    return c.s;
}

DI v8f wmma_bf16(v16bf a, v16bf b, v8f c) {
    // D = A(16x32 bf16) * B(32x16 bf16) + C(16x16 f32)
    return __builtin_amdgcn_wmma_f32_16x16x32_bf16(false, a, false, b,
                                                   (short)0, c, false, false);
}

// Async global->LDS copy, 16B per lane, SADDR form (ASYNCcnt-tracked DMA path).
// vdst = LDS byte offset, voff = 32-bit global byte offset, sbase = uniform base.
DI void async_g2l_b128(unsigned lds_off, unsigned voff, const void* sbase) {
    asm volatile("global_load_async_to_lds_b128 %0, %1, %2"
                 :: "v"(lds_off), "v"(voff), "s"(sbase)
                 : "memory");
}
DI void wait_asynccnt0() {
    asm volatile("s_wait_asynccnt 0" ::: "memory");
}

// pack 16 f32 -> v16bf (hardware cvt, vectorizes to v_cvt_pk_bf16_f32)
DI v16bf pack16(const float* t) {
    BFr r;
#pragma unroll
    for (int j = 0; j < 16; ++j) r.b[j] = (__bf16)t[j];
    return r.v;
}

// A-fragment from f32 memory. base points at row + K-run0 (8 f32); run1 at +16.
// Per-lane K set: lanes 0-15 -> {0..7,16..23}, lanes 16-31 -> {8..15,24..31}.
DI v16bf load_afrag_f32(const float* base) {
    F4 a, b, c, d;
    a.v = *(const float4*)(base +  0);
    b.v = *(const float4*)(base +  4);
    c.v = *(const float4*)(base + 16);
    d.v = *(const float4*)(base + 20);
    float t[16];
#pragma unroll
    for (int j = 0; j < 4; ++j) { t[j] = a.f[j]; t[4+j] = b.f[j]; t[8+j] = c.f[j]; t[12+j] = d.f[j]; }
    return pack16(t);
}

// A-fragment from bf16 memory (two 8-element runs, 16 elements apart)
DI v16bf load_afrag_bf(const unsigned short* base) {
    BFr r;
    r.q[0] = *(const uint4*)(base);
    r.q[1] = *(const uint4*)(base + 16);
    return r.v;
}

// B-fragment: two 8-element bf16 runs, 16 elements apart (col = lane%16, K runs {sel*8..}, {16+sel*8..})
DI v16bf load_bfrag_bf(const unsigned short* p) {
    BFr r;
    r.q[0] = *(const uint4*)(p);
    r.q[1] = *(const uint4*)(p + 16);
    return r.v;
}

// B-fragment from f32: two 8-element runs, 16 elements apart
DI v16bf load_bfrag_f32(const float* p) {
    float t[16];
    F4 a, b, c, d;
    a.v = *(const float4*)(p + 0);  b.v = *(const float4*)(p + 4);
    c.v = *(const float4*)(p + 16); d.v = *(const float4*)(p + 20);
#pragma unroll
    for (int j = 0; j < 4; ++j) { t[j] = a.f[j]; t[4+j] = b.f[j]; t[8+j] = c.f[j]; t[12+j] = d.f[j]; }
    return pack16(t);
}

// ---------------------------------------------------------------------------
// Kernel 0: zero f32 accumulators (kvs, ks_sum, vs_sum) and convert weights
// ---------------------------------------------------------------------------
__global__ void k_prep(float* acc0, const float* wq, const float* wk,
                       unsigned short* wq_bf, unsigned short* wk_bf) {
    int stride = gridDim.x * blockDim.x;
    for (int i = blockIdx.x * blockDim.x + threadIdx.x; i < 66048; i += stride)
        acc0[i] = 0.0f;
    for (int i = blockIdx.x * blockDim.x + threadIdx.x; i < 65536; i += stride) {
        wq_bf[i] = f2bf(wq[i]);
        wk_bf[i] = f2bf(wk[i]);
    }
}

// ---------------------------------------------------------------------------
// Kernel 1: vs_sum[c] = sum_n x[n][c]   (coalesced column reduction)
// ---------------------------------------------------------------------------
__global__ __launch_bounds__(256) void k_vsum(const float* __restrict__ x,
                                              float* __restrict__ vs_sum) {
    int c = threadIdx.x;
    float s = 0.0f;
    for (int n = blockIdx.x; n < NN; n += gridDim.x)
        s += x[(size_t)n * CIN + c];
    atomicAdd(vs_sum + c, s);
}

// ---------------------------------------------------------------------------
// Kernel 2: q/k projection + bias, per-head L2 norm, store qs/ks (bf16),
//           accumulate ks_sum. Weights (256 KB bf16) staged into LDS with
//           GLOBAL_LOAD_ASYNC_TO_LDS_B128 (ASYNCcnt path); biases via ds.
//           One 16-node tile per wave per iteration; q and k computed in
//           sequential phases to keep the live VGPR set under the spill limit.
// ---------------------------------------------------------------------------
__global__ __launch_bounds__(512) void k_proj(const float* __restrict__ x,
                                              const float* __restrict__ wqb,
                                              const float* __restrict__ wkb,
                                              const unsigned short* __restrict__ wq_bf,
                                              unsigned short* __restrict__ qs,
                                              unsigned short* __restrict__ ks,
                                              float* __restrict__ ks_sum) {
    extern __shared__ char smem_raw[];
    unsigned short* Wlds = (unsigned short*)smem_raw;     // [2][256][256] bf16 (Wq then Wk)
    float*          blds = (float*)(Wlds + 131072);       // [2][256] biases

    { // async-DMA the 256 KB contiguous Wq|Wk bf16 block into LDS (16 B/lane)
        const unsigned lds_base = (unsigned)(size_t)(void*)Wlds; // low 32 bits = LDS offset
#pragma unroll
        for (int j = 0; j < 32; ++j) {
            const unsigned off = (threadIdx.x + j * 512) * 16;
            async_g2l_b128(lds_base + off, off, wq_bf);
        }
        for (int i = threadIdx.x; i < 256; i += 512) { blds[i] = wqb[i]; blds[256 + i] = wkb[i]; }
        wait_asynccnt0();
    }
    __syncthreads();

    const int lane = threadIdx.x & 31;
    const int w    = threadIdx.x >> 5;
    const int col  = lane & 15;      // C-fragment column / A-fragment row
    const int sel  = lane >> 4;      // half selector

    float ksacc[16];
#pragma unroll
    for (int ot = 0; ot < 16; ++ot) ksacc[ot] = 0.0f;

    const int gw = blockIdx.x * 16 + w;
    const int nw = gridDim.x * 16;

    for (int tile = gw; tile < NTILES; tile += nw) {
        const int n0 = tile * 16;
        const float* xrow = x + (size_t)(n0 + col) * CIN;

        if (tile + nw < NTILES)
            __builtin_prefetch(x + (size_t)((tile + nw) * 16 + col) * CIN, 0, 1);

        v16bf ax[8];
#pragma unroll
        for (int kc = 0; kc < 8; ++kc)
            ax[kc] = load_afrag_f32(xrow + kc * 32 + sel * 8);

#pragma unroll
        for (int h = 0; h < 4; ++h) {
#pragma unroll
            for (int m = 0; m < 2; ++m) {            // m==0: q phase, m==1: k phase
                const unsigned short* W  = Wlds + m * 65536;
                const float*          bl = blds + m * 256;

                V8 a[4];
#pragma unroll
                for (int t = 0; t < 4; ++t) {
                    const float b = bl[(h * 4 + t) * 16 + col];
#pragma unroll
                    for (int i = 0; i < 8; ++i) a[t].f[i] = b;
                }
                // kc-outer / t-inner: 4 independent WMMA chains per A-fragment
#pragma unroll
                for (int kc = 0; kc < 8; ++kc) {
#pragma unroll
                    for (int t = 0; t < 4; ++t) {
                        const int off = ((h * 4 + t) * 16 + col) * 256 + kc * 32 + sel * 8;
                        a[t].v = wmma_bf16(ax[kc], load_bfrag_bf(W + off), a[t].v);
                    }
                }
                // L2 normalize over the 64 head channels:
                // per-lane partial over the 4 column-tiles, reduce the 16-lane half.
                float rn[8];
#pragma unroll
                for (int i = 0; i < 8; ++i) {
                    float s = 0.0f;
#pragma unroll
                    for (int t = 0; t < 4; ++t) s += a[t].f[i] * a[t].f[i];
                    s += __shfl_xor(s, 1, 32); s += __shfl_xor(s, 2, 32);
                    s += __shfl_xor(s, 4, 32); s += __shfl_xor(s, 8, 32);
                    rn[i] = rsqrtf(s);
                }
                if (m == 0) {
#pragma unroll
                    for (int t = 0; t < 4; ++t)
#pragma unroll
                        for (int i = 0; i < 8; ++i) {
                            const size_t idx =
                                (size_t)(n0 + i + 8 * sel) * HD + h * 64 + t * 16 + col;
                            qs[idx] = f2bf(a[t].f[i] * rn[i]);
                        }
                } else {
#pragma unroll
                    for (int t = 0; t < 4; ++t) {
                        float kssum = 0.0f;
#pragma unroll
                        for (int i = 0; i < 8; ++i) {
                            const float kv = a[t].f[i] * rn[i];
                            const size_t idx =
                                (size_t)(n0 + i + 8 * sel) * HD + h * 64 + t * 16 + col;
                            ks[idx] = f2bf(kv);
                            kssum += kv;
                        }
                        ksacc[h * 4 + t] += kssum;
                    }
                }
            }
        }
    }

    // ks_sum: per-wave register accumulators -> combine halves -> f32 atomics
#pragma unroll
    for (int ot = 0; ot < 16; ++ot) {
        float r = ksacc[ot];
        r += __shfl_xor(r, 16, 32);
        if (lane < 16) atomicAdd(ks_sum + ot * 16 + lane, r);
    }
}

// ---------------------------------------------------------------------------
// Kernel 3: kvs[hm][c] = sum_n ks[n][hm] * x[n][c]  (split-K, WMMA)
// Per 32-node chunk: stage ksT (bf16) and xT (f32) in LDS, build fragments,
// 8 WMMAs per wave, f32 atomic finish. (Transposing stage -> ds stores.)
// ---------------------------------------------------------------------------
__global__ __launch_bounds__(512) void k_kvs(const float* __restrict__ x,
                                             const unsigned short* __restrict__ ks,
                                             float* __restrict__ kvs) {
    extern __shared__ char smem_raw[];
    float*          xT = (float*)smem_raw;                 // [256][36] padded
    unsigned short* kT = (unsigned short*)(xT + 256 * 36); // [256][40] padded

    const int lane = threadIdx.x & 31;
    const int w    = threadIdx.x >> 5;   // wave id 0..15 -> m-tile
    const int col  = lane & 15;
    const int sel  = lane >> 4;

    const int cg    = blockIdx.x & 1;    // c half: tiles cg*8 .. cg*8+7
    const int split = blockIdx.x >> 1;   // split-K index 0..49
    const int ch0 = (NCHUNKS * split) / KVS_SPLIT;
    const int ch1 = (NCHUNKS * (split + 1)) / KVS_SPLIT;

    V8 acc[8];
#pragma unroll
    for (int ct = 0; ct < 8; ++ct)
#pragma unroll
        for (int i = 0; i < 8; ++i) acc[ct].f[i] = 0.0f;

    const int sn  = threadIdx.x & 31;    // staging: node within chunk
    const int seg = threadIdx.x >> 5;    // staging: 16-channel segment

    for (int ch = ch0; ch < ch1; ++ch) {
        const int n0 = ch * 32;
        { // cooperative transpose-stage of this chunk
            const float* xp = x + (size_t)(n0 + sn) * CIN + seg * 16;
            F4 a, b, c, d;
            a.v = *(const float4*)(xp + 0);  b.v = *(const float4*)(xp + 4);
            c.v = *(const float4*)(xp + 8);  d.v = *(const float4*)(xp + 12);
#pragma unroll
            for (int j = 0; j < 4; ++j) {
                xT[(seg * 16 + j     ) * 36 + sn] = a.f[j];
                xT[(seg * 16 + j + 4 ) * 36 + sn] = b.f[j];
                xT[(seg * 16 + j + 8 ) * 36 + sn] = c.f[j];
                xT[(seg * 16 + j + 12) * 36 + sn] = d.f[j];
            }
            BFr kb;
            const uint4* kp = (const uint4*)(ks + (size_t)(n0 + sn) * HD + seg * 16);
            kb.q[0] = kp[0]; kb.q[1] = kp[1];
#pragma unroll
            for (int j = 0; j < 16; ++j)
                kT[(seg * 16 + j) * 40 + sn] = kb.s[j];
        }
        __syncthreads();

        // A fragment: row m = w*16+col of ksT, K-runs {sel*8..} and {16+sel*8..}
        BFr ar;
        const unsigned short* arow = kT + (w * 16 + col) * 40;
        ar.q[0] = *(const uint4*)(arow + sel * 8);
        ar.q[1] = *(const uint4*)(arow + 16 + sel * 8);

#pragma unroll
        for (int ct = 0; ct < 8; ++ct) {
            const float* bp = xT + (size_t)((cg * 8 + ct) * 16 + col) * 36 + sel * 8;
            acc[ct].v = wmma_bf16(ar.v, load_bfrag_f32(bp), acc[ct].v);
        }
        __syncthreads();
    }

    // split-K finish: f32 atomics into kvs
#pragma unroll
    for (int ct = 0; ct < 8; ++ct)
#pragma unroll
        for (int i = 0; i < 8; ++i) {
            const size_t idx = (size_t)(w * 16 + i + 8 * sel) * CIN + (cg * 8 + ct) * 16 + col;
            atomicAdd(kvs + idx, acc[ct].f[i]);
        }
}

// ---------------------------------------------------------------------------
// Kernel 4: out[n][h*256+c] = (qs[n,h,:]·kvs[h,:,c] + vs_sum[c]) /
//                             (qs[n,h,:]·ks_sum[h,:] + N)
// kvs staged transposed in LDS as bf16 B-fragments; qs streamed as A-frags.
// HBM-bound on the 410 MB output write; stores are half-line coalesced.
// ---------------------------------------------------------------------------
__global__ __launch_bounds__(512) void k_out(const unsigned short* __restrict__ qs,
                                             const float* __restrict__ kvs,
                                             const float* __restrict__ ks_sum,
                                             const float* __restrict__ vs_sum,
                                             float* __restrict__ out) {
    extern __shared__ char smem_raw[];
    unsigned short* kvT  = (unsigned short*)smem_raw;      // [256 c][264 hm] bf16
    float*          kss  = (float*)(kvT + 256 * 264);      // 256
    float*          vss  = kss + 256;                      // 256
    float*          nscr = vss + 256;                      // 16 waves * 16 nodes * 4 heads

    { // stage kvs (f32->bf16, transposed) + ks_sum + vs_sum
        const float4* kq = (const float4*)kvs;
        for (int i4 = threadIdx.x; i4 < 16384; i4 += 512) {
            F4 f; f.v = kq[i4];
            const int m = i4 >> 6;
            const int c = (i4 & 63) * 4;
#pragma unroll
            for (int j = 0; j < 4; ++j)
                kvT[(size_t)(c + j) * 264 + m] = f2bf(f.f[j]);
        }
        for (int i = threadIdx.x; i < 256; i += 512) { kss[i] = ks_sum[i]; vss[i] = vs_sum[i]; }
    }
    __syncthreads();

    const int lane = threadIdx.x & 31;
    const int w    = threadIdx.x >> 5;
    const int col  = lane & 15;
    const int sel  = lane >> 4;

    float vsv[16];
#pragma unroll
    for (int ct = 0; ct < 16; ++ct) vsv[ct] = vss[ct * 16 + col];

    const int gw = blockIdx.x * 16 + w;
    const int nw = gridDim.x * 16;

    for (int tile = gw; tile < NTILES; tile += nw) {
        const int n0 = tile * 16;
        const unsigned short* qrow = qs + (size_t)(n0 + col) * HD;

        if (tile + nw < NTILES)
            __builtin_prefetch(qs + (size_t)((tile + nw) * 16 + col) * HD, 0, 1);

        BFr aq[8];
#pragma unroll
        for (int kc = 0; kc < 8; ++kc)
            aq[kc].v = load_afrag_bf(qrow + kc * 32 + sel * 8);

        // normalizer: per-lane partial dot with ks_sum over the lane's K-runs,
        // combined across halves -> full dot for node (lane%16)
#pragma unroll
        for (int h = 0; h < 4; ++h) {
            float p = 0.0f;
#pragma unroll
            for (int kk = 0; kk < 2; ++kk) {
                const int kc = 2 * h + kk;
                const float* kp = kss + kc * 32 + sel * 8;
                BFr& a = aq[kc];
#pragma unroll
                for (int j = 0; j < 8; ++j) {
                    union { unsigned u; float f; } c0, c1;
                    c0.u = (unsigned)a.s[j]     << 16;   // bf16 -> f32 bit-shift
                    c1.u = (unsigned)a.s[8 + j] << 16;
                    p += c0.f * kp[j] + c1.f * kp[16 + j];
                }
            }
            p += __shfl_xor(p, 16, 32);
            if (lane < 16) nscr[w * 64 + lane * 4 + h] = p + 100000.0f;
        }

        float rn[4][8];
#pragma unroll
        for (int h = 0; h < 4; ++h)
#pragma unroll
            for (int i = 0; i < 8; ++i)
                rn[h][i] = 1.0f / nscr[w * 64 + (i + 8 * sel) * 4 + h];

#pragma unroll
        for (int h = 0; h < 4; ++h) {
            for (int ct = 0; ct < 16; ++ct) {
                V8 acc;
#pragma unroll
                for (int i = 0; i < 8; ++i) acc.f[i] = 0.0f;
#pragma unroll
                for (int kk = 0; kk < 2; ++kk) {
                    const int kc = 2 * h + kk;
                    const unsigned short* bp =
                        kvT + (size_t)(ct * 16 + col) * 264 + h * 64 + kk * 32 + sel * 8;
                    acc.v = wmma_bf16(aq[kc].v, load_bfrag_bf(bp), acc.v);
                }
#pragma unroll
                for (int i = 0; i < 8; ++i) {
                    const size_t idx =
                        (size_t)(n0 + i + 8 * sel) * 1024 + h * 256 + ct * 16 + col;
                    out[idx] = (acc.f[i] + vsv[ct]) * rn[h][i];
                }
            }
        }
    }
}

// ---------------------------------------------------------------------------
extern "C" void kernel_launch(void* const* d_in, const int* in_sizes, int n_in,
                              void* d_out, int out_size, void* d_ws, size_t ws_size,
                              hipStream_t stream) {
    (void)in_sizes; (void)n_in; (void)out_size; (void)ws_size;
    const float* x    = (const float*)d_in[0];
    const float* Wq_w = (const float*)d_in[1];
    const float* Wq_b = (const float*)d_in[2];
    const float* Wk_w = (const float*)d_in[3];
    const float* Wk_b = (const float*)d_in[4];
    float* out = (float*)d_out;

    char* ws = (char*)d_ws;
    float*          kvs   = (float*)(ws + KVS_OFF);
    float*          kssum = (float*)(ws + KSSUM_OFF);
    float*          vssum = (float*)(ws + VSSUM_OFF);
    unsigned short* wqbf  = (unsigned short*)(ws + WQ_OFF);  // Wk follows contiguously
    unsigned short* wkbf  = (unsigned short*)(ws + WK_OFF);
    unsigned short* qsb   = (unsigned short*)(ws + QS_OFF);
    unsigned short* ksb   = (unsigned short*)(ws + KS_OFF);

    // 0) zero accumulators + weight bf16 conversion
    k_prep<<<258, 256, 0, stream>>>(kvs, Wq_w, Wk_w, wqbf, wkbf);
    // 1) vs_sum = column sums of x
    k_vsum<<<512, 256, 0, stream>>>(x, vssum);
    // 2) projections + norm + qs/ks + ks_sum  (258 KB LDS; async weight DMA)
    k_proj<<<112, 512, 264192, stream>>>(x, Wq_b, Wk_b, wqbf, qsb, ksb, kssum);
    // 3) kvs = ksT @ x  (split-K=50 x 2 c-halves; 56 KB LDS transpose staging)
    k_kvs<<<2 * KVS_SPLIT, 512, 57344, stream>>>(x, ksb, kvs);
    // 4) output GEMM + epilogue  (~140 KB LDS: kvsT bf16 + sums + norm scratch)
    k_out<<<112, 512, 141312, stream>>>(qsb, kvs, kssum, vssum, out);
}
